// TogepiToeplitz_86354612453666
// MI455X (gfx1250) — matrix operand order repaired and checked by reading, not compiled, not run on hardware
//
#include <hip/hip_runtime.h>
#include <cmath>

#define BATCH 8
#define NUM_HEADS 12
#define DH 64
#define EMBED 768
#define LMAX 1024
#define NTAPS 2047

typedef __attribute__((ext_vector_type(16))) __bf16 v16bf;
typedef __attribute__((ext_vector_type(8)))  __bf16 v8bf;
typedef __attribute__((ext_vector_type(8)))  float  v8f;

static __device__ __forceinline__ v8bf ld8(const __bf16* p) { return *(const v8bf*)p; }
static __device__ __forceinline__ v16bf cat8(v8bf lo, v8bf hi) {
  return __builtin_shufflevector(lo, hi, 0,1,2,3,4,5,6,7,8,9,10,11,12,13,14,15);
}
static __device__ __forceinline__ v8f wmma_bf16(v16bf a, v16bf b, v8f c) {
  return __builtin_amdgcn_wmma_f32_16x16x32_bf16(false, a, false, b, (short)0, c, false, false);
}
// WGP-scope prefetch (scope 0 = pull into all cache levels) — the builtin can
// only reach SE/SYS scopes, so emit the ISA instruction directly.
static __device__ __forceinline__ void prefetch_wgp_2lines(const void* p) {
  asm volatile("global_prefetch_b8 %0, off\n\t"
               "global_prefetch_b8 %0, off offset:128"
               :: "v"(p) : "memory");
}

// ---------------------------------------------------------------------------
// K0: softmax over taps (axis=1) per (h,k) column + causal mask -> d_out tail
// ---------------------------------------------------------------------------
__global__ __launch_bounds__(256) void k_softmax(const float* __restrict__ psfs,
                                                 float* __restrict__ outPsf) {
  __shared__ float red[256];
  const int col = blockIdx.x;            // h*64 + k
  const int h = col >> 6, k = col & 63;
  const size_t base = (size_t)h * NTAPS * DH + k;
  const int tid = threadIdx.x;

  float mx = -3.402823466e38f;
  for (int t = tid; t < NTAPS; t += 256) mx = fmaxf(mx, psfs[base + (size_t)t * DH]);
  red[tid] = mx; __syncthreads();
  for (int s = 128; s > 0; s >>= 1) {
    if (tid < s) red[tid] = fmaxf(red[tid], red[tid + s]);
    __syncthreads();
  }
  mx = red[0]; __syncthreads();

  float sum = 0.f;
  for (int t = tid; t < NTAPS; t += 256) sum += __expf(psfs[base + (size_t)t * DH] - mx);
  red[tid] = sum; __syncthreads();
  for (int s = 128; s > 0; s >>= 1) {
    if (tid < s) red[tid] += red[tid + s];
    __syncthreads();
  }
  const float inv = 1.f / red[0];

  for (int t = tid; t < NTAPS; t += 256) {
    float v = (t < LMAX) ? __expf(psfs[base + (size_t)t * DH] - mx) * inv : 0.f;
    outPsf[base + (size_t)t * DH] = v;
  }
}

// ---------------------------------------------------------------------------
// K1: emb [B,L,D] fp32 -> embT [b][h][n][64] bf16
// ---------------------------------------------------------------------------
__global__ __launch_bounds__(256) void k_cvt_emb(const float* __restrict__ emb,
                                                 __bf16* __restrict__ embT, int total) {
  for (int i = blockIdx.x * blockDim.x + threadIdx.x; i < total; i += gridDim.x * blockDim.x) {
    const int b = i / (LMAX * EMBED);
    const int r = i % (LMAX * EMBED);
    const int n = r / EMBED;
    const int d = r % EMBED;
    const int h = d >> 6, j = d & 63;
    embT[((size_t)(b * NUM_HEADS + h) * LMAX + n) * DH + j] = (__bf16)emb[i];
  }
}

// ---------------------------------------------------------------------------
// K2: proj_w fp32 -> bf16 row-major [o][c] (= WMMA B [K,N] source for y@W^T)
// ---------------------------------------------------------------------------
__global__ __launch_bounds__(256) void k_cvt_w(const float* __restrict__ w,
                                               __bf16* __restrict__ o) {
  const int i = blockIdx.x * blockDim.x + threadIdx.x;
  if (i < EMBED * EMBED) o[i] = (__bf16)w[i];
}

// ---------------------------------------------------------------------------
// K3: expand masked psf rows into WMMA-B-fragment-ready circulant blobs:
// blob[h][t][frag=dblk*2+c][lane][16 bf16], value = psf[h,t,(d-j)&63],
// d = dblk*16 + (lane&15), j = c*32 + (lane>=16)*8 + v + (v>=8)*8  (dense-B layout).
// ---------------------------------------------------------------------------
__global__ __launch_bounds__(256) void k_expand(const float* __restrict__ psf,
                                                __bf16* __restrict__ blob) {
  __shared__ float row[DH];
  const int tid = threadIdx.x;
  if (tid < DH) {
    const int h = blockIdx.x / LMAX;
    const int t = blockIdx.x % LMAX;
    row[tid] = psf[((size_t)h * NTAPS + t) * DH + tid];
  }
  __syncthreads();
  const int f = tid >> 5, lane = tid & 31;
  const int d  = ((f >> 1) << 4) | (lane & 15);
  const int jb = ((f & 1) << 5) | ((lane >> 4) << 3);
  v16bf w;
#pragma unroll
  for (int v = 0; v < 16; ++v) w[v] = (__bf16)row[(d - (jb + v + ((v >> 3) << 3))) & 63];
  *(v16bf*)(blob + ((size_t)blockIdx.x * 8 + f) * 512 + (size_t)lane * 16) = w;
}

// ---------------------------------------------------------------------------
// K4: causal Toeplitz conv via WMMA.  One workgroup = (b, h, 32-row pair).
// A fragments stream from L2-resident embT; each 8KB circulant tap feeds
// 16 WMMAs (2 row subtiles).  4 waves split the tap blocks; only the two
// boundary blocks need m>=0 guards and run as epilogues.  WGP-scope inline
// asm prefetch 4 taps ahead; cross-wave reduce in 32KB LDS; fused GELU.
// ---------------------------------------------------------------------------
__global__ __launch_bounds__(128) void k_conv(const __bf16* __restrict__ embT,
                                              const __bf16* __restrict__ blob,
                                              __bf16* __restrict__ y) {
  __shared__ float ldsRed[4 * 2 * 4 * 8 * 32];   // [wave][sub][dblk][v][lane] 32KB

  const int p = 31 - (blockIdx.x & 31);          // descending p: heavy blocks first
  const int h = (blockIdx.x >> 5) % NUM_HEADS;
  const int b = blockIdx.x / (32 * NUM_HEADS);
  const int n0 = p << 5;
  const int tid = threadIdx.x, wid = tid >> 5, lane = tid & 31;
  const int r = lane & 15;
  const int half8 = (lane >> 4) << 3;

  const __bf16* aBase = embT + (size_t)(b * NUM_HEADS + h) * LMAX * DH;
  const __bf16* bbase = blob + (size_t)h * LMAX * 4096 + (size_t)lane * 16;
  const __bf16* pfb   = blob + (size_t)h * LMAX * 4096 + (size_t)lane * 128;

  v8f acc[2][4] = {};

  // ---- main loop: tap blocks 0..2p-1, fully in-range, branch-free --------
  for (int tb = wid; tb < 2 * p; tb += 4) {
    const int t0 = tb << 4;
    const __bf16* row0 = aBase + (size_t)(n0 + r - t0) * DH + half8;
    const __bf16* bf = bbase + (size_t)t0 * 4096;
    // prefetch pointer marches 4 taps ahead of bf (lane-strided, 2 lines
    // cover the full 8KB tap blob); blob is not the last ws region, so the
    // <=32KB tail overrun of the last head stays inside the workspace.
    const __bf16* pf = pfb + (size_t)(t0 + 4) * 4096;
#pragma unroll 1
    for (int tt = 0; tt < 16; ++tt) {
      v16bf a00 = cat8(ld8(row0),      ld8(row0 + 16));
      v16bf a01 = cat8(ld8(row0 + 32), ld8(row0 + 48));
      const __bf16* row1 = row0 + 16 * DH;
      v16bf a10 = cat8(ld8(row1),      ld8(row1 + 16));
      v16bf a11 = cat8(ld8(row1 + 32), ld8(row1 + 48));
      prefetch_wgp_2lines(pf);
#pragma unroll
      for (int dblk = 0; dblk < 4; ++dblk) {
        v16bf b0 = *(const v16bf*)(bf + dblk * 1024);
        v16bf b1 = *(const v16bf*)(bf + dblk * 1024 + 512);
        acc[0][dblk] = wmma_bf16(a00, b0, acc[0][dblk]);
        acc[1][dblk] = wmma_bf16(a10, b0, acc[1][dblk]);
        acc[0][dblk] = wmma_bf16(a01, b1, acc[0][dblk]);
        acc[1][dblk] = wmma_bf16(a11, b1, acc[1][dblk]);
      }
      row0 -= DH;            // t -> t+1  => source row m -> m-1
      bf   += 4096;
      pf   += 4096;
    }
  }

  // ---- boundary block tb = 2p (t = n0+tt): subtile0 guarded, subtile1 full
  if (wid == ((2 * p) & 3)) {
#pragma unroll 1
    for (int tt = 0; tt < 16; ++tt) {
      const int t = n0 + tt;
      const int m0 = r - tt;                      // abs source row, subtile0
      v16bf a00 = {}, a01 = {};
      if (m0 >= 0) {
        const __bf16* row0 = aBase + (size_t)m0 * DH + half8;
        a00 = cat8(ld8(row0),      ld8(row0 + 16));
        a01 = cat8(ld8(row0 + 32), ld8(row0 + 48));
      }
      const __bf16* row1 = aBase + (size_t)(16 + r - tt) * DH + half8; // >=1
      v16bf a10 = cat8(ld8(row1),      ld8(row1 + 16));
      v16bf a11 = cat8(ld8(row1 + 32), ld8(row1 + 48));
      const __bf16* bf = bbase + (size_t)t * 4096;
#pragma unroll
      for (int dblk = 0; dblk < 4; ++dblk) {
        v16bf b0 = *(const v16bf*)(bf + dblk * 1024);
        v16bf b1 = *(const v16bf*)(bf + dblk * 1024 + 512);
        acc[0][dblk] = wmma_bf16(a00, b0, acc[0][dblk]);
        acc[1][dblk] = wmma_bf16(a10, b0, acc[1][dblk]);
        acc[0][dblk] = wmma_bf16(a01, b1, acc[0][dblk]);
        acc[1][dblk] = wmma_bf16(a11, b1, acc[1][dblk]);
      }
    }
  }

  // ---- boundary block tb = 2p+1 (t = n0+16+tt): only subtile1, guarded ---
  if (wid == ((2 * p + 1) & 3)) {
#pragma unroll 1
    for (int tt = 0; tt < 16; ++tt) {
      const int t = n0 + 16 + tt;
      const int m1 = r - tt;                      // abs source row, subtile1
      v16bf a10 = {}, a11 = {};
      if (m1 >= 0) {
        const __bf16* row1 = aBase + (size_t)m1 * DH + half8;
        a10 = cat8(ld8(row1),      ld8(row1 + 16));
        a11 = cat8(ld8(row1 + 32), ld8(row1 + 48));
      }
      const __bf16* bf = bbase + (size_t)t * 4096;
#pragma unroll
      for (int dblk = 0; dblk < 4; ++dblk) {
        v16bf b0 = *(const v16bf*)(bf + dblk * 1024);
        v16bf b1 = *(const v16bf*)(bf + dblk * 1024 + 512);
        acc[1][dblk] = wmma_bf16(a10, b0, acc[1][dblk]);
        acc[1][dblk] = wmma_bf16(a11, b1, acc[1][dblk]);
      }
    }
  }

  // ---- cross-wave reduction, GELU, store ---------------------------------
#pragma unroll
  for (int s = 0; s < 2; ++s)
#pragma unroll
    for (int dblk = 0; dblk < 4; ++dblk)
#pragma unroll
      for (int v = 0; v < 8; ++v)
        ldsRed[(((wid * 2 + s) * 4 + dblk) * 8 + v) * 32 + lane] = acc[s][dblk][v];
  __syncthreads();

  const size_t ybase = ((size_t)b * LMAX + n0) * EMBED + h * DH;
#pragma unroll
  for (int i = 0; i < 16; ++i) {
    const int e = tid * 16 + i;                 // 0..2047 over 32x64 outputs
    const int nl = e >> 6, d = e & 63;
    const int s = nl >> 4, nls = nl & 15;
    const int dblk = d >> 4;
    const int sl = (d & 15) | ((nls >> 3) << 4);
    const int v = nls & 7;
    float acc_s = 0.f;
#pragma unroll
    for (int w = 0; w < 4; ++w)
      acc_s += ldsRed[(((w * 2 + s) * 4 + dblk) * 8 + v) * 32 + sl];
    acc_s = 0.5f * acc_s * (1.0f + erff(acc_s * 0.70710678118654752f));
    y[ybase + (size_t)nl * EMBED + d] = (__bf16)acc_s;
  }
}

// ---------------------------------------------------------------------------
// K5: out[8192,768] = y @ proj_w^T + b.  One wave = 32x64 tile (B reused x2).
// ---------------------------------------------------------------------------
__global__ __launch_bounds__(128) void k_proj(const __bf16* __restrict__ y,
                                              const __bf16* __restrict__ wp,
                                              const float* __restrict__ bias,
                                              float* __restrict__ out) {
  const int tid = threadIdx.x, wid = tid >> 5, lane = tid & 31;
  const int tile = blockIdx.x * 4 + wid;    // 0..3071
  const int rowPair = tile & 255;           // 256 row pairs of 32
  const int colTile = tile >> 8;            // 12 col tiles of 64
  const int i0 = rowPair << 5;
  const int o0 = colTile << 6;
  const int r = lane & 15;
  const int half8  = (lane >> 4) << 3;
  const int half16 = (lane >> 4) << 4;

  v8f acc[2][4] = {};
  const __bf16* arow0 = y + (size_t)(i0 + r) * EMBED;
  const __bf16* arow1 = arow0 + 16 * EMBED;
#pragma unroll 1
  for (int kb = 0; kb < EMBED; kb += 32) {
    v16bf a0 = cat8(ld8(arow0 + kb + half16), ld8(arow0 + kb + half16 + 8));
    v16bf a1 = cat8(ld8(arow1 + kb + half16), ld8(arow1 + kb + half16 + 8));
#pragma unroll
    for (int ot = 0; ot < 4; ++ot) {
      const __bf16* brow = wp + (size_t)(o0 + ot * 16 + r) * EMBED + kb + half16;
      v16bf bb = *(const v16bf*)brow;
      acc[0][ot] = wmma_bf16(a0, bb, acc[0][ot]);
      acc[1][ot] = wmma_bf16(a1, bb, acc[1][ot]);
    }
  }

  const int nOff = (lane >> 4) << 3;
#pragma unroll
  for (int s = 0; s < 2; ++s)
#pragma unroll
    for (int ot = 0; ot < 4; ++ot) {
      const int o = o0 + ot * 16 + (lane & 15);
      const float bv = bias[o];
#pragma unroll
      for (int v = 0; v < 8; ++v) {
        const int irow = i0 + s * 16 + v + nOff;
        out[(size_t)irow * EMBED + o] = acc[s][ot][v] + bv;
      }
    }
}

// ---------------------------------------------------------------------------
extern "C" void kernel_launch(void* const* d_in, const int* in_sizes, int n_in,
                              void* d_out, int out_size, void* d_ws, size_t ws_size,
                              hipStream_t stream) {
  const float* emb  = (const float*)d_in[0];   // [8,1024,768]
  const float* psfs = (const float*)d_in[1];   // [12,2047,64]
  const float* pw   = (const float*)d_in[2];   // [768,768]
  const float* pb   = (const float*)d_in[3];   // [768]

  float* out    = (float*)d_out;
  float* outPsf = out + (size_t)BATCH * LMAX * EMBED;

  // Workspace layout (blob deliberately NOT last so the prefetch window's
  // tail overrun stays inside the allocation):
  char* ws = (char*)d_ws;
  __bf16* embT = (__bf16*)(ws);                   //  12,582,912 B  [b][h][n][64]
  __bf16* wp   = (__bf16*)(ws + 12582912);        //   1,179,648 B  [o][c]
  __bf16* blob = (__bf16*)(ws + 13762560);        // 100,663,296 B  circulant frags
  __bf16* yb   = (__bf16*)(ws + 114425856);       //  12,582,912 B  [b*n][768]

  k_softmax<<<NUM_HEADS * DH, 256, 0, stream>>>(psfs, outPsf);
  k_cvt_emb<<<2048, 256, 0, stream>>>(emb, embT, BATCH * LMAX * EMBED);
  k_cvt_w<<<(EMBED * EMBED + 255) / 256, 256, 0, stream>>>(pw, wp);
  k_expand<<<NUM_HEADS * LMAX, 256, 0, stream>>>(outPsf, blob);
  k_conv<<<BATCH * NUM_HEADS * 32, 128, 0, stream>>>(embT, blob, yb);
  k_proj<<<768, 128, 0, stream>>>(yb, wp, pb, out);
}
